// HandModel_35579509080129
// MI455X (gfx1250) — matrix operand order repaired and checked by reading, not compiled, not run on hardware
//
#include <hip/hip_runtime.h>
#include <hip/hip_bf16.h>

typedef unsigned short u16;
typedef __attribute__((ext_vector_type(16))) __bf16 v16bf;
typedef __attribute__((ext_vector_type(8)))  float  v8f;

#define B_    16
#define T_    24
#define HIN   112
#define HP    56
#define WP    56
#define F_    32
#define NPIX  (B_*HP*WP)       // 50176
#define KTOT  320              // 288 (Uk) + 27 (Wk) + 5 pad
#define MTILE 64
#define PHW   58               // padded h: 58x58
#define DROW  (HP*WP*F_)       // 100352

__device__ __forceinline__ u16 f2bf(float f) {
    unsigned u = __float_as_uint(f);
    u += 0x7FFFu + ((u >> 16) & 1u);
    return (u16)(u >> 16);
}
__device__ __forceinline__ float hsig(float z) {
    return fminf(fmaxf(0.2f * z + 0.5f, 0.0f), 1.0f);
}

// ---- zero h double-buffers (incl. halo) and cell state, every call ----
__global__ void zero_ws_kernel(u16* h0, u16* h1, float* c, int nh, int nc) {
    int i = blockIdx.x * blockDim.x + threadIdx.x;
    int st = gridDim.x * blockDim.x;
    for (int j = i; j < nh; j += st) { h0[j] = 0; h1[j] = 0; }
    for (int j = i; j < nc; j += st) { c[j] = 0.0f; }
}

// ---- pack Uk (3,3,32,128) + Wk (3,3,3,128) fp32 -> UkT_ext[128][320] bf16 ----
__global__ void prep_ukt_kernel(const float* __restrict__ Uk,
                                const float* __restrict__ Wk,
                                u16* __restrict__ ukt) {
    int idx = blockIdx.x * blockDim.x + threadIdx.x;   // 128*320 = 40960
    if (idx >= 128 * KTOT) return;
    int n = idx / KTOT, k = idx - n * KTOT;
    float v = 0.0f;
    if (k < 288) {                       // recurrent taps: k = tap*32 + c
        int tap = k >> 5, c = k & 31;
        v = Uk[(size_t)(tap * 32 + c) * 128 + n];
    } else if (k < 315) {                // input taps: k-288 = tap*3 + ci
        int r = k - 288;
        int tap = r / 3, ci = r - tap * 3;
        v = Wk[(size_t)(tap * 3 + ci) * 128 + n];
    }
    ukt[(size_t)n * KTOT + k] = f2bf(v);
}

// ---- one ConvLSTM time step: fused (Wk*x + Uk*h) GEMM + gates ----
__global__ __launch_bounds__(256) void step_kernel(
    const u16* __restrict__ hin, u16* __restrict__ hout,
    float* __restrict__ cst, const u16* __restrict__ ukt,
    const float* __restrict__ bias, const float* __restrict__ x,
    float* __restrict__ out, int t)
{
    __shared__ __align__(16) unsigned char smem[MTILE * KTOT * 2];  // 40 KB
    u16*   sA = (u16*)smem;     // im2col tile: [64][320] bf16
    float* sZ = (float*)smem;   // z tile: [64][128] f32 (aliased)

    const int tid = threadIdx.x;
    const int m0  = blockIdx.x * MTILE;

    // ---- im2col, recurrent part: 64 px * 9 taps, 64B contiguous chunks ----
    for (int idx = tid; idx < MTILE * 9; idx += 256) {
        int p = idx / 9, tap = idx - p * 9;
        int gp  = m0 + p;
        int img = gp / (HP * WP);
        int pix = gp - img * (HP * WP);
        int oy = pix / WP, ox = pix - oy * WP;
        int dy = tap / 3,  dx = tap - dy * 3;
        const int4* src = (const int4*)(hin +
            (size_t)(((img * PHW + oy + dy) * PHW) + ox + dx) * F_);
        int4* dst = (int4*)(sA + p * KTOT + tap * F_);
        dst[0] = src[0]; dst[1] = src[1]; dst[2] = src[2]; dst[3] = src[3];
    }
    // ---- im2col, input-conv part (stride 2, SAME: pad_begin 0) + K pad ----
    for (int p = tid; p < MTILE; p += 256) {
        int gp  = m0 + p;
        int img = gp / (HP * WP);
        int pix = gp - img * (HP * WP);
        int oy = pix / WP, ox = pix - oy * WP;
        const float* xb = x + (size_t)(img * T_ + t) * HIN * HIN * 3;
        u16* row = sA + p * KTOT + 288;
        #pragma unroll
        for (int tap = 0; tap < 9; ++tap) {
            int dy = tap / 3, dx = tap - dy * 3;
            int iy = 2 * oy + dy, ix = 2 * ox + dx;
            bool ok = (iy < HIN) && (ix < HIN);
            const float* s = xb + (size_t)(iy * HIN + ix) * 3;
            #pragma unroll
            for (int ci = 0; ci < 3; ++ci)
                row[tap * 3 + ci] = ok ? f2bf(s[ci]) : (u16)0;
        }
        row[27] = 0; row[28] = 0; row[29] = 0; row[30] = 0; row[31] = 0;
    }
    __syncthreads();

    // ---- implicit GEMM: 4 Mtiles x 1 Ntile per wave, K = 10 blocks of 32 ----
    const int lane = tid & 31;
    const int w    = tid >> 5;                 // wave -> channel tile 16w..16w+15
    const int hi   = (lane >= 16) ? 1 : 0;
    const int ln   = lane & 15;
    v8f acc[MTILE / 16] = {};
    const u16* bcol = ukt + (size_t)(w * 16 + ln) * KTOT + hi * 16;
    #pragma unroll
    for (int kb = 0; kb < KTOT / 32; ++kb) {
        v16bf bfrag = *(const v16bf*)(bcol + kb * 32);
        #pragma unroll
        for (int mt = 0; mt < MTILE / 16; ++mt) {
            const u16* ap = sA + (mt * 16 + ln) * KTOT + kb * 32 + hi * 16;
            struct A32 { int4 a, b; } av;
            av.a = *(const int4*)ap;
            av.b = *(const int4*)(ap + 8);
            v16bf afrag = __builtin_bit_cast(v16bf, av);
            acc[mt] = __builtin_amdgcn_wmma_f32_16x16x32_bf16(
                false, afrag, false, bfrag, (short)0, acc[mt], false, false);
        }
    }
    __syncthreads();   // all waves done reading sA before aliasing as sZ

    // ---- scatter accumulators to LDS z-tile [64][128] ----
    #pragma unroll
    for (int mt = 0; mt < MTILE / 16; ++mt) {
        #pragma unroll
        for (int j = 0; j < 8; ++j) {
            int p = mt * 16 + j + hi * 8;
            sZ[p * 128 + w * 16 + ln] = acc[mt][j];
        }
    }
    __syncthreads();

    // ---- gates: i,f,c,o per (pixel, channel) ----
    for (int e = tid; e < MTILE * F_; e += 256) {
        int p = e >> 5, ch = e & 31;
        int gp  = m0 + p;
        int img = gp / (HP * WP);
        int pix = gp - img * (HP * WP);
        float zi = sZ[p * 128 + ch]      + bias[ch];
        float zf = sZ[p * 128 + ch + 32] + bias[ch + 32];
        float zc = sZ[p * 128 + ch + 64] + bias[ch + 64];
        float zo = sZ[p * 128 + ch + 96] + bias[ch + 96];
        float ig = hsig(zi), fg = hsig(zf), og = hsig(zo);
        size_t cidx = (size_t)gp * F_ + ch;
        float cp = cst[cidx];
        float cn = fg * cp + ig * tanhf(zc);
        float hn = og * tanhf(cn);
        cst[cidx] = cn;
        int oy = pix / WP, ox = pix - oy * WP;
        hout[(size_t)(((img * PHW + oy + 1) * PHW) + ox + 1) * F_ + ch] = f2bf(hn);
        out[(size_t)(img * T_ + t) * DROW + (size_t)pix * F_ + ch] = hn;
    }
}

// ---- LayerNorm over D per (B,T) row, in place ----
__global__ __launch_bounds__(256) void ln_kernel(float* __restrict__ y,
                                                 const float* __restrict__ gamma,
                                                 const float* __restrict__ beta) {
    __shared__ float s1[256], s2[256];
    float* row = y + (size_t)blockIdx.x * DROW;
    float sum = 0.0f, sq = 0.0f;
    for (int i = threadIdx.x; i < DROW; i += 256) {
        float v = row[i]; sum += v; sq += v * v;
    }
    s1[threadIdx.x] = sum; s2[threadIdx.x] = sq;
    __syncthreads();
    for (int off = 128; off > 0; off >>= 1) {
        if (threadIdx.x < off) {
            s1[threadIdx.x] += s1[threadIdx.x + off];
            s2[threadIdx.x] += s2[threadIdx.x + off];
        }
        __syncthreads();
    }
    float mu  = s1[0] * (1.0f / DROW);
    float var = s2[0] * (1.0f / DROW) - mu * mu;
    float inv = rsqrtf(var + 1e-3f);
    for (int i = threadIdx.x; i < DROW; i += 256)
        row[i] = (row[i] - mu) * inv * gamma[i] + beta[i];
}

extern "C" void kernel_launch(void* const* d_in, const int* in_sizes, int n_in,
                              void* d_out, int out_size, void* d_ws, size_t ws_size,
                              hipStream_t stream) {
    (void)in_sizes; (void)n_in; (void)out_size; (void)ws_size;
    const float* x     = (const float*)d_in[0];   // (16,24,112,112,3)
    const float* Wk    = (const float*)d_in[1];   // (3,3,3,128)
    const float* Uk    = (const float*)d_in[2];   // (3,3,32,128)
    const float* bias  = (const float*)d_in[3];   // (128,)
    const float* gamma = (const float*)d_in[4];   // (100352,)
    const float* beta  = (const float*)d_in[5];   // (100352,)
    float* out = (float*)d_out;                   // (16,24,100352)

    char* ws = (char*)d_ws;
    size_t off = 0;
    auto alloc = [&](size_t bytes) -> void* {
        void* p = (void*)(ws + off);
        off += (bytes + 255) & ~(size_t)255;
        return p;
    };
    const size_t NH = (size_t)B_ * PHW * PHW * F_;      // 1,722,368 bf16 elems
    const size_t NC = (size_t)B_ * HP * WP * F_;        // 1,605,632 f32 elems
    u16*   ukt = (u16*)  alloc((size_t)128 * KTOT * 2); // 80 KB
    u16*   h0  = (u16*)  alloc(NH * 2);
    u16*   h1  = (u16*)  alloc(NH * 2);
    float* cst = (float*)alloc(NC * 4);

    zero_ws_kernel<<<2048, 256, 0, stream>>>(h0, h1, cst, (int)NH, (int)NC);
    prep_ukt_kernel<<<(128 * KTOT + 255) / 256, 256, 0, stream>>>(Uk, Wk, ukt);

    for (int t = 0; t < T_; ++t) {
        const u16* hin = (t & 1) ? h1 : h0;
        u16*      hout = (t & 1) ? h0 : h1;
        step_kernel<<<NPIX / MTILE, 256, 0, stream>>>(
            hin, hout, cst, ukt, bias, x, out, t);
    }
    ln_kernel<<<B_ * T_, 256, 0, stream>>>(out, gamma, beta);
}
